// PeriodicConvolution_21191368639122
// MI455X (gfx1250) — matrix operand order repaired and checked by reading, not compiled, not run on hardware
//
#include <hip/hip_runtime.h>
#include <hip/hip_bf16.h>

// CDNA5 / gfx1250: wave32, WMMA 16x16x32 bf16 path.
typedef __attribute__((ext_vector_type(16))) __bf16 v16bf;
typedef __attribute__((ext_vector_type(8)))  float  v8f;

#define B_    4
#define P_    8192
#define E_    262144     // 2^18
#define CIN   32
#define COUT  32
#define H_    16
#define GAMMA_ 4.0f

#define THREADS 256
#define BLOCKS  1024

__global__ __launch_bounds__(THREADS) void pc_zero_kernel(float4* out4, int n4) {
    int i = blockIdx.x * blockDim.x + threadIdx.x;
    if (i < n4) out4[i] = make_float4(0.f, 0.f, 0.f, 0.f);
}

__global__ __launch_bounds__(THREADS) void pc_conv_kernel(
    const float* __restrict__ features,   // [B,P,CIN]
    const float* __restrict__ edge_vec,   // [B,E,3]
    const float* __restrict__ W,          // [H,COUT,CIN]
    const float* __restrict__ mu,         // [H]
    const int*   __restrict__ edge_src,   // [B,E]
    const int*   __restrict__ edge_dst,   // [B,E]
    const int*   __restrict__ n_norm_p,   // [1]
    float*       __restrict__ out)        // [B,P,COUT]
{
    // W packed as bf16 WMMA A-fragments: fragment (h, tile) is the 16x32 tile
    // A[m,k] = W[h, o = tile*16+m, i = k], stored per-lane in contiguous k order (k = 16*hiK + j).
    __shared__ __attribute__((aligned(32))) __bf16 Wlds[H_ * 2 * 32 * 16]; // 32 KB

    const int tid = threadIdx.x;

    // ---- cooperative pack: fp32 W -> bf16 A-fragment order in LDS ----
    // 16-bit A 16x32 layout: lane = m + 16*hiK ; half j (0..15):
    //   k = j + hiK*16
    for (int f = tid; f < H_ * 2 * 32 * 16; f += THREADS) {
        int j    = f & 15;
        int lane = (f >> 4) & 31;
        int t    = (f >> 9) & 1;
        int h    = f >> 10;
        int m    = lane & 15;
        int hiK  = lane >> 4;
        int k    = j + hiK * 16;
        int o    = t * 16 + m;
        Wlds[f]  = (__bf16)W[(h * COUT + o) * CIN + k];
    }
    __syncthreads();

    const int lane = tid & 31;
    const int n    = lane & 15;   // this lane's edge within the 16-edge group
    const int hi   = lane >> 4;   // this lane's K-half (channels hi*16..hi*16+15)
    const float scale = rsqrtf((float)n_norm_p[0]);

    float mur[H_];
    #pragma unroll
    for (int h = 0; h < H_; ++h) mur[h] = mu[h];

    const int num_groups = (B_ * E_) >> 4;                 // 65536 groups of 16 edges
    const int wave  = (blockIdx.x * THREADS + tid) >> 5;
    const int nwave = (gridDim.x * THREADS) >> 5;

    const v16bf* wfrag = (const v16bf*)Wlds;

    for (int g = wave; g < num_groups; g += nwave) {
        const int fe = (g << 4) + n;       // flat edge index in [0, B*E)
        const int b  = fe >> 18;           // E = 2^18

        // prefetch next group's edge stream (linear; speculative, safe OOB)
        {
            const int feN = ((g + nwave) << 4) + n;
            __builtin_prefetch(edge_vec + (size_t)feN * 3, 0, 0);
            __builtin_prefetch(edge_src + feN, 0, 0);
        }

        // ---- per-edge geometry + RBF (both K-half lanes recompute for edge n) ----
        const float* ev = edge_vec + (size_t)fe * 3;
        float vx = ev[0], vy = ev[1], vz = ev[2];
        float r = sqrtf(vx * vx + vy * vy + vz * vz);
        r = (r < 1e-10f) ? 0.0f : r;       // self-interaction zeroing

        const int src = edge_src[fe];
        const int dst = edge_dst[fe];

        // 16 RBF values, converted once to packed bf16 (8 regs, v_cvt_pk_bf16_f32)
        v16bf rbfv;
        #pragma unroll
        for (int h = 0; h < H_; ++h) {
            float d = r - mur[h];
            rbfv[h] = (__bf16)__expf(-GAMMA_ * d * d);   // v_exp_f32
        }

        // ---- gather this lane's 16-channel slice of x[src] (B-fragment K-slice) ----
        const float* xrow = features + ((size_t)b * P_ + src) * CIN + hi * 16;
        float xf[16];
        #pragma unroll
        for (int j = 0; j < 16; j += 4) {
            float4 t4 = *(const float4*)(xrow + j);
            xf[j]   = t4.x; xf[j+1] = t4.y;
            xf[j+2] = t4.z; xf[j+3] = t4.w;
        }
        v16bf xbf;
        #pragma unroll
        for (int j = 0; j < 16; ++j) xbf[j] = (__bf16)xf[j];

        // ---- K-loop: one RBF basis per 16x16x32 WMMA step, 2 output tiles ----
        // Fully unrolled 2-stage software pipeline with three pinned sub-regions
        // per step:  [wmma pair] | [A-frag ds preloads (h+2)] | [bfrag pk_muls
        // (h+1)].  The ds_load issues sit between the WMMAs and the pk_muls so
        // the WMMA->VALU WAR hazard (pk_mul reusing WMMA-read VGPRs) has 4
        // intervening issue slots -> no v_nop padding; LDS latency stays
        // hidden 2 steps deep; liveness stays bounded (~5 fragment buffers).
        v8f acc0 = {0.f,0.f,0.f,0.f,0.f,0.f,0.f,0.f};
        v8f acc1 = {0.f,0.f,0.f,0.f,0.f,0.f,0.f,0.f};

        v16bf a0b[2], a1b[2], bfr[2];
        a0b[0] = wfrag[0 * 32 + lane];     // (h=0, t=0)
        a1b[0] = wfrag[1 * 32 + lane];     // (h=0, t=1)
        a0b[1] = wfrag[2 * 32 + lane];     // (h=1, t=0)
        a1b[1] = wfrag[3 * 32 + lane];     // (h=1, t=1)
        {
            __bf16 rb0 = rbfv[0];
            v16bf rv0;
            #pragma unroll
            for (int j = 0; j < 16; ++j) rv0[j] = rb0;
            bfr[0] = xbf * rv0;            // bfrag for h=0
        }

        #pragma unroll
        for (int h = 0; h < H_; ++h) {
            const int p = h & 1;

            acc0 = __builtin_amdgcn_wmma_f32_16x16x32_bf16(
                       false, a0b[p], false, bfr[p], (short)0, acc0, false, false);
            acc1 = __builtin_amdgcn_wmma_f32_16x16x32_bf16(
                       false, a1b[p], false, bfr[p], (short)0, acc1, false, false);
            __builtin_amdgcn_sched_barrier(0);

            if (h + 2 < H_) {                          // A-fragments for h+2
                a0b[p] = wfrag[((h + 2) * 2 + 0) * 32 + lane];
                a1b[p] = wfrag[((h + 2) * 2 + 1) * 32 + lane];
            }
            __builtin_amdgcn_sched_barrier(0);

            if (h + 1 < H_) {                          // bfrag for h+1 (other buffer)
                __bf16 rb = rbfv[h + 1];
                v16bf rv;
                #pragma unroll
                for (int j = 0; j < 16; ++j) rv[j] = rb;
                bfr[p ^ 1] = xbf * rv;                 // v_pk_mul_bf16 x8
            }
            __builtin_amdgcn_sched_barrier(0);         // end of pipeline stage
        }

        // ---- scatter: D[m,n] lives in the lane owning edge n; o = tile*16 + hi*8 + q
        float* op = out + ((size_t)b * P_ + dst) * COUT + hi * 8;
        #pragma unroll
        for (int q = 0; q < 8; ++q)
            unsafeAtomicAdd(op + q,      acc0[q] * scale);   // global_atomic_add_f32
        #pragma unroll
        for (int q = 0; q < 8; ++q)
            unsafeAtomicAdd(op + 16 + q, acc1[q] * scale);
    }
}

extern "C" void kernel_launch(void* const* d_in, const int* in_sizes, int n_in,
                              void* d_out, int out_size, void* d_ws, size_t ws_size,
                              hipStream_t stream) {
    const float* features = (const float*)d_in[0];
    const float* edge_vec = (const float*)d_in[1];
    const float* W        = (const float*)d_in[2];
    const float* mu       = (const float*)d_in[3];
    const int*   edge_src = (const int*)d_in[4];
    const int*   edge_dst = (const int*)d_in[5];
    const int*   n_norm   = (const int*)d_in[6];
    float* out = (float*)d_out;

    // zero the accumulation target (out_size = B*P*COUT = 1,048,576 floats)
    int n4 = out_size / 4;
    pc_zero_kernel<<<(n4 + THREADS - 1) / THREADS, THREADS, 0, stream>>>((float4*)out, n4);

    pc_conv_kernel<<<BLOCKS, THREADS, 0, stream>>>(
        features, edge_vec, W, mu, edge_src, edge_dst, n_norm, out);
}
